// MoeLayer_27668179320917
// MI455X (gfx1250) — matrix operand order repaired and checked by reading, not compiled, not run on hardware
//
#include <hip/hip_runtime.h>
#include <cstddef>

// MoE sizes (fixed by the reference)
#define TT 1024
#define DD 2048
#define HH 7168
#define EE 8

typedef __attribute__((ext_vector_type(16))) __bf16   v16bf;
typedef __attribute__((ext_vector_type(8)))  float    v8f;
typedef __attribute__((ext_vector_type(4)))  unsigned v4u;
typedef __attribute__((ext_vector_type(8)))  unsigned v8u;

// ---------------------------------------------------------------------------
// Tensor Data Mover: issue a 2D tile load (fp32 elements) Global -> LDS.
// D# per CDNA5 ISA ch.8: group0 = {count/flags, lds_addr, global_addr, type=2},
// group1 = {data_size, tensor dims, tile dims, dim0 stride}.  One instruction
// per wave (EXEC ignored); tracked with TENSORcnt.
__device__ inline unsigned lds_addr_of(const void* p) {
  // generic LDS pointer: addr[31:0] is the LDS byte address (ISA 10.2)
  return (unsigned)(unsigned long long)p;
}

__device__ inline void tdm_load_2d_f32(unsigned lds_addr, const void* gaddr,
                                       unsigned tile_x, unsigned tile_y,
                                       unsigned stride_elems) {
  unsigned long long ga = (unsigned long long)gaddr;
  const unsigned tens_x = 1u << 20, tens_y = 1u << 20;  // generous OOB dims
  v4u g0;
  g0[0] = 1u;                                             // count=1, user D#
  g0[1] = lds_addr;                                       // LDS byte address
  g0[2] = (unsigned)ga;                                   // global_addr[31:0]
  g0[3] = (unsigned)((ga >> 32) & 0x01ffffffu) | (2u << 30);  // [56:32]|type=2
  v8u g1;
  g1[0] = 2u << 16;                                       // data_size=4B, mask=0
  g1[1] = (tens_x & 0xffffu) << 16;                       // tensor_dim0[15:0]
  g1[2] = ((tens_x >> 16) & 0xffffu) | ((tens_y & 0xffffu) << 16);
  g1[3] = ((tens_y >> 16) & 0xffffu) | ((tile_x & 0xffffu) << 16);
  g1[4] = (tile_y & 0xffffu);                             // tile_dim1, dim2=0
  g1[5] = stride_elems;                                   // dim0_stride[31:0]
  g1[6] = 0u;                                             // stride hi, dim1_stride
  g1[7] = 0u;
  asm volatile("tensor_load_to_lds %0, %1" :: "s"(g0), "s"(g1) : "memory");
}

// ---------------------------------------------------------------------------
// Fragment loaders (wave32 WMMA 16x16x32 bf16 layouts, ISA 7.12.2)
// A 16x32 (MxK), bf16 row-major in LDS.
__device__ inline v16bf load_a_frag(const __bf16* As, int lane) {
  int m  = lane & 15;
  int kb = (lane < 16) ? 0 : 16;
  union { v16bf v; unsigned u[8]; } r;
  const unsigned* row = (const unsigned*)(As + m * 32);
#pragma unroll
  for (int j = 0; j < 4; ++j) {
    r.u[j]     = row[(kb >> 1) + j];
    r.u[4 + j] = row[(kb >> 1) + 4 + j];
  }
  return r.v;
}

// B 32x16 (KxN) from fp32 LDS tile stored [n][32] (TDM row order), cvt->bf16.
__device__ inline v16bf load_b_frag_f32(const float* Bs, int lane) {
  int n  = lane & 15;
  int kb = (lane < 16) ? 0 : 16;
  const float* p = Bs + n * 32 + kb;
  union { v16bf v; __bf16 b[16]; } r;
#pragma unroll
  for (int j = 0; j < 16; ++j) r.b[j] = (__bf16)p[j];
  return r.v;
}

// B 32x16 (KxN) from fp32 LDS tile stored [k][128] (TDM row order for w2).
__device__ inline v16bf load_b_frag_f32T(const float* Bs, int col, int lane) {
  int kb = (lane < 16) ? 0 : 16;
  union { v16bf v; __bf16 b[16]; } r;
#pragma unroll
  for (int j = 0; j < 16; ++j) r.b[j] = (__bf16)Bs[(kb + j) * 128 + col];
  return r.v;
}

// ---------------------------------------------------------------------------
__global__ __launch_bounds__(256) void moe_zero(float* __restrict__ out, int n,
                                                int* __restrict__ counts) {
  int i = blockIdx.x * 256 + threadIdx.x;
  if (i < n) out[i] = 0.0f;
  if (i < EE) counts[i] = 0;
}

// ---------------------------------------------------------------------------
// Router: one wave per token, top-2 + softmax, append to per-expert lists.
__global__ __launch_bounds__(256) void moe_router(const float* __restrict__ x,
                                                  const float* __restrict__ gw,
                                                  int*   __restrict__ counts,
                                                  int*   __restrict__ list_tok,
                                                  float* __restrict__ list_w) {
  int tid  = threadIdx.x;
  int lane = tid & 31;
  int t    = blockIdx.x * 8 + (tid >> 5);

  float acc[EE];
#pragma unroll
  for (int e = 0; e < EE; ++e) acc[e] = 0.0f;

  const float* xr = x + (size_t)t * DD;
  for (int d = lane; d < DD; d += 32) {
    float xv = xr[d];
#pragma unroll
    for (int e = 0; e < EE; ++e) acc[e] += xv * gw[(size_t)e * DD + d];
  }
#pragma unroll
  for (int e = 0; e < EE; ++e)
#pragma unroll
    for (int off = 16; off; off >>= 1) acc[e] += __shfl_xor(acc[e], off, 32);

  if (lane == 0) {
    int e0 = 0; float v0 = acc[0];
#pragma unroll
    for (int e = 1; e < EE; ++e) if (acc[e] > v0) { v0 = acc[e]; e0 = e; }
    int e1 = -1; float v1 = -3.4e38f;
#pragma unroll
    for (int e = 0; e < EE; ++e)
      if (e != e0 && acc[e] > v1) { v1 = acc[e]; e1 = e; }
    float w0 = __builtin_amdgcn_rcpf(1.0f + __expf(v1 - v0));
    float w1 = 1.0f - w0;
    int p0 = atomicAdd(&counts[e0], 1);
    list_tok[e0 * TT + p0] = (t << 1) | 0;  list_w[e0 * TT + p0] = w0;
    int p1 = atomicAdd(&counts[e1], 1);
    list_tok[e1 * TT + p1] = (t << 1) | 1;  list_w[e1 * TT + p1] = w1;
  }
}

// ---------------------------------------------------------------------------
// FFN part 1: h = silu(x.w1^T) * (x.w3^T) for the gathered tokens of expert e.
// grid = (E, H/128, T/16), block = 256 (8 waves, one 16x16 N-subtile each).
// Weight tiles (128 rows x 32 k fp32) come in via TDM; A is a 16-row gather.
__global__ __launch_bounds__(256) void moe_ffn1(const float* __restrict__ x,
                                                const float* __restrict__ w1,
                                                const float* __restrict__ w3,
                                                const int*   __restrict__ counts,
                                                const int*   __restrict__ list_tok,
                                                unsigned short* __restrict__ h_ws) {
  __shared__ __bf16 As[16 * 32];
  __shared__ float  B1f[128 * 32];
  __shared__ float  B3f[128 * 32];
  __shared__ int    entries[16];

  const int e   = blockIdx.x;
  const int n0  = blockIdx.y * 128;
  const int tb  = blockIdx.z * 16;
  const int cnt = counts[e];
  if (tb >= cnt) return;

  const int tid  = threadIdx.x;
  const int lane = tid & 31;
  const int wid  = tid >> 5;

  if (tid < 16) {
    int p = tb + tid;
    entries[tid] = (p < cnt) ? list_tok[e * TT + p] : 0;
  }
  __syncthreads();

  // Hoisted gather pointer for this thread's A-staging slot (loop-invariant).
  const int arow = tid >> 4, acol = (tid & 15) * 2;
  const float* xrow = x + (size_t)(entries[arow] >> 1) * DD + acol;

  const float* w1base = w1 + ((size_t)e * HH + n0) * DD;
  const float* w3base = w3 + ((size_t)e * HH + n0) * DD;
  const unsigned lds_b1 = lds_addr_of(B1f);
  const unsigned lds_b3 = lds_addr_of(B3f);

  v8f accg = {};
  v8f accu = {};

  for (int k0 = 0; k0 < DD; k0 += 32) {
    __syncthreads();
    if (wid == 0) {                       // one TDM issue per workgroup
      tdm_load_2d_f32(lds_b1, w1base + k0, 32, 128, DD);
      tdm_load_2d_f32(lds_b3, w3base + k0, 32, 128, DD);
    }
    // Stage A (overlaps the tensor DMA): 16 gathered x rows -> bf16
    {
      float2 v = *(const float2*)(xrow + k0);
      As[arow * 32 + acol]     = (__bf16)v.x;
      As[arow * 32 + acol + 1] = (__bf16)v.y;
    }
    if (wid == 0) __builtin_amdgcn_s_wait_tensorcnt(0);
    __syncthreads();

    v16bf a  = load_a_frag(As, lane);
    v16bf b1 = load_b_frag_f32(B1f + wid * 16 * 32, lane);
    v16bf b3 = load_b_frag_f32(B3f + wid * 16 * 32, lane);
    accg = __builtin_amdgcn_wmma_f32_16x16x32_bf16(false, a, false, b1,
                                                   (short)0, accg, false, false);
    accu = __builtin_amdgcn_wmma_f32_16x16x32_bf16(false, a, false, b3,
                                                   (short)0, accu, false, false);
  }

  // Epilogue: h = silu(g)*u, store bf16 to h_ws[(token<<1|slot)][n]
  __bf16* h = (__bf16*)h_ws;
#pragma unroll
  for (int r = 0; r < 8; ++r) {
    int m = r + ((lane < 16) ? 0 : 8);
    int col = lane & 15;
    if (tb + m < cnt) {
      float g = accg[r];
      float u = accu[r];
      float hv = g * __builtin_amdgcn_rcpf(1.0f + __expf(-g)) * u;
      h[(size_t)entries[m] * HH + n0 + wid * 16 + col] = (__bf16)hv;
    }
  }
}

// ---------------------------------------------------------------------------
// FFN part 2: y = h . w2[e]; out[token] += weight * y (scatter-add).
// grid = (E, D/128, T/16), block = 256.  w2 tile (32 k x 128 d fp32) via TDM.
__global__ __launch_bounds__(256) void moe_ffn2(const unsigned short* __restrict__ h_ws,
                                                const float* __restrict__ w2,
                                                const int*   __restrict__ counts,
                                                const int*   __restrict__ list_tok,
                                                const float* __restrict__ list_w,
                                                float* __restrict__ out) {
  __shared__ __bf16 As[16 * 32];
  __shared__ float  Bf[32 * 128];          // TDM row order: [k][d]
  __shared__ int    entries[16];
  __shared__ float  wts[16];

  const int e   = blockIdx.x;
  const int d0  = blockIdx.y * 128;
  const int tb  = blockIdx.z * 16;
  const int cnt = counts[e];
  if (tb >= cnt) return;

  const int tid  = threadIdx.x;
  const int lane = tid & 31;
  const int wid  = tid >> 5;

  if (tid < 16) {
    int p = tb + tid;
    entries[tid] = (p < cnt) ? list_tok[e * TT + p] : 0;
    wts[tid]     = (p < cnt) ? list_w[e * TT + p] : 0.0f;
  }
  __syncthreads();

  const __bf16* h = (const __bf16*)h_ws;
  // Hoisted gather pointer for this thread's A-staging slot.
  const int arow = tid >> 4, apair = tid & 15;
  const unsigned* hrow =
      (const unsigned*)(h + (size_t)entries[arow] * HH) + apair;

  const float* w2base = w2 + (size_t)e * HH * DD + d0;
  const unsigned lds_b = lds_addr_of(Bf);
  const int mycol = wid * 16 + (lane & 15);

  v8f acc = {};

  for (int k0 = 0; k0 < HH; k0 += 32) {
    __syncthreads();
    if (wid == 0)
      tdm_load_2d_f32(lds_b, w2base + (size_t)k0 * DD, 128, 32, DD);
    // Stage A: 16 h rows (bf16), one dword per thread
    ((unsigned*)As)[arow * 16 + apair] = hrow[k0 >> 1];
    if (wid == 0) __builtin_amdgcn_s_wait_tensorcnt(0);
    __syncthreads();

    v16bf a = load_a_frag(As, lane);
    v16bf b = load_b_frag_f32T(Bf, mycol, lane);
    acc = __builtin_amdgcn_wmma_f32_16x16x32_bf16(false, a, false, b,
                                                  (short)0, acc, false, false);
  }

#pragma unroll
  for (int r = 0; r < 8; ++r) {
    int m = r + ((lane < 16) ? 0 : 8);
    int col = lane & 15;
    if (tb + m < cnt) {
      int tok = entries[m] >> 1;
      atomicAdd(&out[(size_t)tok * DD + d0 + wid * 16 + col], wts[m] * acc[r]);
    }
  }
}

// ---------------------------------------------------------------------------
extern "C" void kernel_launch(void* const* d_in, const int* in_sizes, int n_in,
                              void* d_out, int out_size, void* d_ws, size_t ws_size,
                              hipStream_t stream) {
  (void)in_sizes; (void)n_in; (void)out_size; (void)ws_size;

  const float* x  = (const float*)d_in[0];
  const float* gw = (const float*)d_in[1];
  const float* w1 = (const float*)d_in[2];
  const float* w2 = (const float*)d_in[3];
  const float* w3 = (const float*)d_in[4];
  float* out = (float*)d_out;

  char* ws = (char*)d_ws;
  int*   counts   = (int*)ws;                                  // [E]
  int*   list_tok = (int*)(ws + 256);                          // [E][T]
  float* list_w   = (float*)(ws + 256 + EE * TT * 4);          // [E][T]
  unsigned short* h_ws = (unsigned short*)(ws + (1 << 17));    // [2T][H] bf16

  moe_zero<<<(TT * DD + 255) / 256, 256, 0, stream>>>(out, TT * DD, counts);
  moe_router<<<TT / 8, 256, 0, stream>>>(x, gw, counts, list_tok, list_w);
  moe_ffn1<<<dim3(EE, HH / 128, TT / 16), 256, 0, stream>>>(
      x, w1, w3, counts, list_tok, h_ws);
  moe_ffn2<<<dim3(EE, DD / 128, TT / 16), 256, 0, stream>>>(
      h_ws, w2, counts, list_tok, list_w, out);
}
